// AttentionLayer_37778532335764
// MI455X (gfx1250) — matrix operand order repaired and checked by reading, not compiled, not run on hardware
//
#include <hip/hip_runtime.h>
#include <hip/hip_bf16.h>

// ---------------------------------------------------------------------------
// CDNA5 (gfx1250) attention layer, all GEMMs via v_wmma_f32_16x16x32_bf16,
// with async global->LDS staging (GLOBAL_LOAD_ASYNC_TO_LDS_B128 / ASYNCcnt):
//   Wq/Wk/Wv : f32 -> bf16 + transpose (GEMM B operands always N x K)
//   Q = q@Wq, K = k@Wk         (bf16 out, row-major)
//   Vt = (v@Wv)^T              (bf16 out, transposed store)
//   S = Q@K^T                  (f32 out)
//   P = softmax(S/sqrt(dk))    (bf16 out)
//   O = P@V  (B = Vt, N x K)   (f32 out -> d_out)
// ---------------------------------------------------------------------------

typedef __attribute__((ext_vector_type(16))) __bf16 v16bf;
typedef __attribute__((ext_vector_type(8)))  float  v8f;
typedef __attribute__((ext_vector_type(4)))  unsigned int u32x4;
typedef __attribute__((ext_vector_type(4)))  float  f32x4;

#define TILE_M 128
#define TILE_N 128
#define TILE_K 32
#define LDS_STRIDE 40   // 32 + 8 bf16 pad to stagger LDS banks

union FragU  { u32x4 u[2]; v16bf v; };
union PackU  { __bf16 h[8]; u32x4 u; };

// Async copy of one 32-byte run (two b128) from global to LDS.
// GVS mode: saddr = uniform base (SGPR pair), vaddr = per-lane 32-bit byte
// offset, INST_OFFSET applies to BOTH the LDS and global addresses.
__device__ __forceinline__
void async_ldst32(unsigned lds_addr, const __bf16* base, unsigned voff) {
    asm volatile(
        "global_load_async_to_lds_b128 %0, %1, %2\n\t"
        "global_load_async_to_lds_b128 %0, %1, %2 offset:16"
        :: "v"(lds_addr), "v"(voff), "s"(base) : "memory");
}
__device__ __forceinline__ void async_wait0() {
    asm volatile("s_wait_asynccnt 0x0" ::: "memory");
}

// --------------------------- f32 -> bf16 convert ---------------------------
__global__ __launch_bounds__(256)
void cvt_f32_bf16(const float* __restrict__ in, __bf16* __restrict__ out, int n4) {
    int i = blockIdx.x * blockDim.x + threadIdx.x;
    int stride = gridDim.x * blockDim.x;
    for (; i < n4; i += stride) {
        f32x4 x = ((const f32x4*)in)[i];
        __bf16* o = out + (size_t)i * 4;
        o[0] = (__bf16)x.x; o[1] = (__bf16)x.y;
        o[2] = (__bf16)x.z; o[3] = (__bf16)x.w;
    }
}

// -------------------- f32 -> bf16 convert + transpose ----------------------
// in: R x C (f32, row-major)   out: C x R (bf16, row-major)
__global__ __launch_bounds__(256)
void cvt_t_f32_bf16(const float* __restrict__ in, __bf16* __restrict__ out,
                    int R, int C) {
    __shared__ __bf16 t[32][33];
    const int bc = blockIdx.x * 32;
    const int br = blockIdx.y * 32;
    const int tc = threadIdx.x & 31;
    const int tr0 = threadIdx.x >> 5;
    #pragma unroll
    for (int i = 0; i < 4; ++i) {
        int r = tr0 + i * 8;
        t[tc][r] = (__bf16)in[(size_t)(br + r) * C + bc + tc];
    }
    __syncthreads();
    #pragma unroll
    for (int i = 0; i < 4; ++i) {
        int r = tr0 + i * 8;
        out[(size_t)(bc + r) * R + br + tc] = t[r][tc];
    }
}

// ------------------------------- tiled GEMM --------------------------------
// C(MxN) = A(MxK) * B^T where B is (N x K) row-major bf16.
// 256 threads = 8 waves (2 in M x 4 in N). Per wave: 64x32 output = 8 WMMA accs.
// Double-buffered LDS filled by the async global->LDS engine (ASYNCcnt).
template<bool OUT_BF16, bool OUT_T>
__global__ __launch_bounds__(256)
void gemm_bf16(const __bf16* __restrict__ A, const __bf16* __restrict__ B,
               void* __restrict__ Cout, int M, int N, int K) {
    __shared__ __bf16 sA[2][TILE_M * LDS_STRIDE];
    __shared__ __bf16 sB[2][TILE_N * LDS_STRIDE];

    const int tid  = threadIdx.x;
    const int bm   = blockIdx.y * TILE_M;
    const int bn   = blockIdx.x * TILE_N;
    const int wave = tid >> 5;
    const int lane = tid & 31;
    const int wm   = (wave & 1) * 64;   // 2 wave-rows of 64
    const int wn   = (wave >> 1) * 32;  // 4 wave-cols of 32
    const int row16 = lane & 15;
    const int half  = lane >> 4;

    v8f acc[4][2] = {};

    // staging map: thread t copies one 32-byte run: row = t>>1, col = (t&1)*16
    const int srow = tid >> 1;
    const int scol = (tid & 1) * 16;
    const __bf16* baseA = A + (size_t)bm * K;   // uniform SGPR base
    const __bf16* baseB = B + (size_t)bn * K;
    const unsigned voff0 = (unsigned)((srow * K + scol) * 2);   // bytes
    const unsigned ldsA[2] = { (unsigned)(uintptr_t)&sA[0][srow * LDS_STRIDE + scol],
                               (unsigned)(uintptr_t)&sA[1][srow * LDS_STRIDE + scol] };
    const unsigned ldsB[2] = { (unsigned)(uintptr_t)&sB[0][srow * LDS_STRIDE + scol],
                               (unsigned)(uintptr_t)&sB[1][srow * LDS_STRIDE + scol] };

    const int nk = K / TILE_K;

    // ---- preload tile 0 via async engine ----
    async_ldst32(ldsA[0], baseA, voff0);
    async_ldst32(ldsB[0], baseB, voff0);
    async_wait0();
    __syncthreads();

    for (int kt = 0; kt < nk; ++kt) {
        const int cur = kt & 1;

        // ---- kick off async copy of tile kt+1 (overlaps the WMMAs) ----
        if (kt + 1 < nk) {
            const unsigned voff = voff0 + (unsigned)(kt + 1) * (TILE_K * 2);
            async_ldst32(ldsA[cur ^ 1], baseA, voff);
            async_ldst32(ldsB[cur ^ 1], baseB, voff);
        }

        // ---- fragments from LDS (documented CDNA5 wave32 layouts) ----
        const __bf16* sa = &sA[cur][0];
        const __bf16* sb = &sB[cur][0];
        FragU a[4], b0, b1;
        #pragma unroll
        for (int i = 0; i < 4; ++i) {
            const int r = (wm + i * 16 + row16) * LDS_STRIDE;
            a[i].u[0] = *(const u32x4*)&sa[r + half * 8];
            a[i].u[1] = *(const u32x4*)&sa[r + 16 + half * 8];
        }
        {
            const int r0 = (wn + row16) * LDS_STRIDE;
            const int r1 = (wn + 16 + row16) * LDS_STRIDE;
            b0.u[0] = *(const u32x4*)&sb[r0 + half * 8];
            b0.u[1] = *(const u32x4*)&sb[r0 + 16 + half * 8];
            b1.u[0] = *(const u32x4*)&sb[r1 + half * 8];
            b1.u[1] = *(const u32x4*)&sb[r1 + 16 + half * 8];
        }
        #pragma unroll
        for (int i = 0; i < 4; ++i) {
            acc[i][0] = __builtin_amdgcn_wmma_f32_16x16x32_bf16(
                false, a[i].v, false, b0.v, (short)0, acc[i][0], false, false);
            acc[i][1] = __builtin_amdgcn_wmma_f32_16x16x32_bf16(
                false, a[i].v, false, b1.v, (short)0, acc[i][1], false, false);
        }

        // ---- drain async engine, publish tile kt+1 ----
        if (kt + 1 < nk) {
            async_wait0();
            __syncthreads();
        }
    }

    // ---- epilogue: element (m = r + 8*half, n = lane%16) lives in VGPR r ----
    #pragma unroll
    for (int i = 0; i < 4; ++i) {
        const int cm = bm + wm + i * 16 + half * 8;
        const int cn = bn + wn + row16;
        if constexpr (OUT_T) {
            // C is N x M bf16: 8 consecutive m per lane -> one 16B store
            __bf16* C = (__bf16*)Cout;
            PackU p0, p1;
            #pragma unroll
            for (int r = 0; r < 8; ++r) {
                p0.h[r] = (__bf16)acc[i][0][r];
                p1.h[r] = (__bf16)acc[i][1][r];
            }
            *(u32x4*)&C[(size_t)cn * M + cm]        = p0.u;
            *(u32x4*)&C[(size_t)(cn + 16) * M + cm] = p1.u;
        } else if constexpr (OUT_BF16) {
            __bf16* C = (__bf16*)Cout;
            #pragma unroll
            for (int r = 0; r < 8; ++r) {
                C[(size_t)(cm + r) * N + cn]      = (__bf16)acc[i][0][r];
                C[(size_t)(cm + r) * N + cn + 16] = (__bf16)acc[i][1][r];
            }
        } else {
            float* C = (float*)Cout;
            #pragma unroll
            for (int r = 0; r < 8; ++r) {
                C[(size_t)(cm + r) * N + cn]      = acc[i][0][r];
                C[(size_t)(cm + r) * N + cn + 16] = acc[i][1][r];
            }
        }
    }
}

// ------------------------------ row softmax --------------------------------
__global__ __launch_bounds__(256)
void softmax_row(const float* __restrict__ S, __bf16* __restrict__ P,
                 int N, float scale) {
    __shared__ float red[8];
    const int row = blockIdx.x;
    const int tid = threadIdx.x;
    const float* s = S + (size_t)row * N;

    float m = -3.4e38f;
    for (int i = tid; i < N; i += 256) m = fmaxf(m, s[i] * scale);
    #pragma unroll
    for (int off = 16; off; off >>= 1) m = fmaxf(m, __shfl_xor(m, off));
    if ((tid & 31) == 0) red[tid >> 5] = m;
    __syncthreads();
    m = red[0];
    #pragma unroll
    for (int i = 1; i < 8; ++i) m = fmaxf(m, red[i]);
    __syncthreads();

    float sum = 0.f;
    for (int i = tid; i < N; i += 256) sum += __expf(s[i] * scale - m);
    #pragma unroll
    for (int off = 16; off; off >>= 1) sum += __shfl_xor(sum, off);
    if ((tid & 31) == 0) red[tid >> 5] = sum;
    __syncthreads();
    sum = 0.f;
    #pragma unroll
    for (int i = 0; i < 8; ++i) sum += red[i];
    const float inv = 1.0f / sum;

    __bf16* p = P + (size_t)row * N;
    for (int i = tid; i < N; i += 256)
        p[i] = (__bf16)(__expf(s[i] * scale - m) * inv);
}

// ------------------------------- launcher ----------------------------------
extern "C" void kernel_launch(void* const* d_in, const int* in_sizes, int n_in,
                              void* d_out, int out_size, void* d_ws, size_t ws_size,
                              hipStream_t stream) {
    const int NQ = 4096, NKV = 4096, DIM = 1024, DK = 1024, DV = 1024;

    const float* q  = (const float*)d_in[0];
    const float* k  = (const float*)d_in[1];
    const float* v  = (const float*)d_in[2];
    const float* Wq = (const float*)d_in[3];
    const float* Wk = (const float*)d_in[4];
    const float* Wv = (const float*)d_in[5];

    // ---- workspace layout ----
    char* w = (char*)d_ws;
    __bf16* qb  = (__bf16*)w; w += (size_t)NQ  * DIM * 2;
    __bf16* kb  = (__bf16*)w; w += (size_t)NKV * DIM * 2;
    __bf16* vb  = (__bf16*)w; w += (size_t)NKV * DIM * 2;
    __bf16* Wqt = (__bf16*)w; w += (size_t)DK  * DIM * 2;
    __bf16* Wkt = (__bf16*)w; w += (size_t)DV  * DIM * 2;
    __bf16* Wvt = (__bf16*)w; w += (size_t)DV  * DIM * 2;
    __bf16* Qb  = (__bf16*)w; w += (size_t)NQ  * DK  * 2;
    __bf16* Kb  = (__bf16*)w; w += (size_t)NKV * DK  * 2;
    __bf16* Vt  = (__bf16*)w; w += (size_t)DV  * NKV * 2;
    float*  Sf  = (float*)w;  w += (size_t)NQ  * NKV * 4;
    __bf16* Pb  = (__bf16*)w; w += (size_t)NQ  * NKV * 2;

    // ---- 1) convert activations; convert+transpose weights ----
    cvt_f32_bf16<<<2048, 256, 0, stream>>>(q, qb, NQ  * DIM / 4);
    cvt_f32_bf16<<<2048, 256, 0, stream>>>(k, kb, NKV * DIM / 4);
    cvt_f32_bf16<<<2048, 256, 0, stream>>>(v, vb, NKV * DIM / 4);
    dim3 gT(DK / 32, DIM / 32);
    cvt_t_f32_bf16<<<gT, 256, 0, stream>>>(Wq, Wqt, DIM, DK);
    cvt_t_f32_bf16<<<gT, 256, 0, stream>>>(Wk, Wkt, DIM, DV);
    cvt_t_f32_bf16<<<gT, 256, 0, stream>>>(Wv, Wvt, DIM, DV);

    // ---- 2) projections ----
    dim3 gProj(DK / TILE_N, NQ / TILE_M);
    gemm_bf16<true, false><<<gProj, 256, 0, stream>>>(qb, Wqt, Qb, NQ,  DK, DIM);
    gemm_bf16<true, false><<<gProj, 256, 0, stream>>>(kb, Wkt, Kb, NKV, DV, DIM);
    gemm_bf16<true, true ><<<gProj, 256, 0, stream>>>(vb, Wvt, Vt, NKV, DV, DIM);

    // ---- 3) scores: S = Q @ K^T ----
    dim3 gS(NKV / TILE_N, NQ / TILE_M);
    gemm_bf16<false, false><<<gS, 256, 0, stream>>>(Qb, Kb, Sf, NQ, NKV, DK);

    // ---- 4) softmax (folds 1/sqrt(dk)) ----
    softmax_row<<<NQ, 256, 0, stream>>>(Sf, Pb, NKV, 0.03125f);

    // ---- 5) O = P @ V  (B = Vt, N x K) ----
    dim3 gO(DV / TILE_N, NQ / TILE_M);
    gemm_bf16<false, false><<<gO, 256, 0, stream>>>(Pb, Vt, (float*)d_out, NQ, DV, NKV);
}
